// HGNNLayer_62620623175867
// MI455X (gfx1250) — matrix operand order repaired and checked by reading, not compiled, not run on hardware
//
#include <hip/hip_runtime.h>
#include <hip/hip_bf16.h>
#include <hip/hip_fp16.h>

typedef __attribute__((ext_vector_type(16))) _Float16 v16h;
typedef __attribute__((ext_vector_type(8)))  _Float16 v8h;
typedef __attribute__((ext_vector_type(8)))  float    v8f;

#define B_  64
#define U_  64
#define T_  512
#define H_  8
#define KNN 9
#define NEGV (-1000000000.0f)

union V16 { v16h v; v8h h[2]; _Float16 e[16]; };
union V8F { v8f v; float f[8]; };

__device__ __forceinline__ float sigmoidf_(float x) { return 1.0f / (1.0f + expf(-x)); }
__device__ __forceinline__ float eluf_(float x)     { return x > 0.0f ? x : expm1f(x); }

// ---------------------------------------------------------------------------
// K1: embeddings + projections.
//   task rows: task_embed = xW_task+b ; tk = eW_K+b_K ; tq = eW_V+b_V
//   usv  rows: usv_embed  = xW_usv+b  ; uq = eW_Q+b_Q
//   Also stores task_embed transposed in f16: teT[b][h][t] (WMMA B operand).
// ---------------------------------------------------------------------------
__global__ void k_embed(const float* __restrict__ usv_f, const float* __restrict__ task_f,
                        const float* __restrict__ W_usv, const float* __restrict__ b_usv,
                        const float* __restrict__ W_task, const float* __restrict__ b_task,
                        const float* __restrict__ W_Q, const float* __restrict__ b_Q,
                        const float* __restrict__ W_K, const float* __restrict__ b_K,
                        const float* __restrict__ W_V, const float* __restrict__ b_V,
                        float* __restrict__ task_embed, float* __restrict__ tk,
                        float* __restrict__ tq, float* __restrict__ usv_embed,
                        float* __restrict__ uq, _Float16* __restrict__ teT)
{
    int idx = blockIdx.x * blockDim.x + threadIdx.x;
    if (idx < B_ * T_) {
        int b = idx / T_, t = idx % T_;
        const float* x = task_f + (size_t)idx * 5;
        float e[H_];
        for (int h = 0; h < H_; ++h) {
            float a = b_task[h];
            for (int j = 0; j < 5; ++j) a += x[j] * W_task[j * H_ + h];
            e[h] = a;
        }
        for (int h = 0; h < H_; ++h) {
            task_embed[(size_t)idx * H_ + h] = e[h];
            teT[((size_t)b * H_ + h) * T_ + t] = (_Float16)e[h];
        }
        for (int h = 0; h < H_; ++h) {
            float ak = b_K[h], av = b_V[h];
            for (int g = 0; g < H_; ++g) {
                ak += e[g] * W_K[g * H_ + h];
                av += e[g] * W_V[g * H_ + h];
            }
            tk[(size_t)idx * H_ + h] = ak;
            tq[(size_t)idx * H_ + h] = av;
        }
    } else if (idx < B_ * T_ + B_ * U_) {
        int i = idx - B_ * T_;
        const float* x = usv_f + (size_t)i * 4;
        float e[H_];
        for (int h = 0; h < H_; ++h) {
            float a = b_usv[h];
            for (int j = 0; j < 4; ++j) a += x[j] * W_usv[j * H_ + h];
            e[h] = a;
        }
        for (int h = 0; h < H_; ++h) usv_embed[(size_t)i * H_ + h] = e[h];
        for (int h = 0; h < H_; ++h) {
            float aq = b_Q[h];
            for (int g = 0; g < H_; ++g) aq += e[g] * W_Q[g * H_ + h];
            uq[(size_t)i * H_ + h] = aq;
        }
    }
}

// ---------------------------------------------------------------------------
// K2: usv->task attention weights. One wave32 per (b,u): each lane owns 16 t's.
// Softmax over T with adj mask; normalized weights stored as f16 (WMMA A op).
// ---------------------------------------------------------------------------
__global__ void k_usv_attn(const float* __restrict__ uq, const float* __restrict__ tk,
                           const int* __restrict__ adj,
                           const float* __restrict__ W_au, const float* __restrict__ b_au,
                           _Float16* __restrict__ w16)
{
    int lane = threadIdx.x & 31;
    int row  = blockIdx.x * (blockDim.x >> 5) + (threadIdx.x >> 5); // b*U+u
    if (row >= B_ * U_) return;
    int b = row / U_;

    float q[H_], wau[H_];
    for (int h = 0; h < H_; ++h) { q[h] = uq[(size_t)row * H_ + h]; wau[h] = W_au[h]; }
    float bau = b_au[0];

    const float* tkb  = tk + (size_t)b * T_ * H_;
    const int*   adjr = adj + (size_t)row * T_;

    float sv[16];
    float mx = -3.4e38f;
    for (int it = 0; it < 16; ++it) {
        int t = lane + 32 * it;
        float acc = bau;
        for (int h = 0; h < H_; ++h) acc += tanhf(q[h] + tkb[(size_t)t * H_ + h]) * wau[h];
        float s = (adjr[t] == 0) ? NEGV : acc;
        sv[it] = s;
        mx = fmaxf(mx, s);
    }
    for (int m = 16; m >= 1; m >>= 1) mx = fmaxf(mx, __shfl_xor(mx, m, 32));
    float sum = 0.0f;
    for (int it = 0; it < 16; ++it) { float p = expf(sv[it] - mx); sv[it] = p; sum += p; }
    for (int m = 16; m >= 1; m >>= 1) sum += __shfl_xor(sum, m, 32);
    float inv = 1.0f / sum;

    _Float16* wr = w16 + (size_t)row * T_;
    for (int it = 0; it < 16; ++it) wr[lane + 32 * it] = (_Float16)(sv[it] * inv);
}

// ---------------------------------------------------------------------------
// K3: agg_task = w(64x512) @ task_embed(512x8) via v_wmma_f32_16x16x32_f16.
// One wave per (b, 16-row tile); fused epilogue: usv_updated = elu(sig(uq)*emb+agg).
// B columns 8..15 are fed duplicated valid data (n&7): they only influence
// D columns 8..15, which are never stored -> no divergence in the K loop.
// ---------------------------------------------------------------------------
__global__ void k_aggtask_wmma(const _Float16* __restrict__ w16, const _Float16* __restrict__ teT,
                               const float* __restrict__ uq, const float* __restrict__ usv_embed,
                               float* __restrict__ out_usv, _Float16* __restrict__ usvuT)
{
    int lane  = threadIdx.x & 31;
    int mtile = threadIdx.x >> 5;          // block = 128 threads = 4 waves
    int b     = blockIdx.x;
    int half  = lane >> 4;
    int n     = lane & 15;                 // N col for B/C/D; also M row-in-tile for A

    V16 a, bb;
    V8F c;
    for (int i = 0; i < 8; ++i) c.f[i] = 0.0f;

    const _Float16* Arow = w16 + ((size_t)(b * U_ + mtile * 16 + n)) * T_;
    const _Float16* Brow = teT + ((size_t)b * H_ + (n & 7)) * T_;

    for (int kk = 0; kk < T_ / 32; ++kk) {
        int k0 = kk * 32;
        // A 16x32 f16 layout: lanes 0-15 K=0..7,16..23 ; lanes 16-31 K=8..15,24..31
        a.h[0] = *(const v8h*)(Arow + k0 + 8 * half);
        a.h[1] = *(const v8h*)(Arow + k0 + 16 + 8 * half);
        // B 32x16 f16 layout: lanes 0-15 K=0..7,16..23 ; lanes 16-31 K=8..15,24..31
        bb.h[0] = *(const v8h*)(Brow + k0 + 8 * half);
        bb.h[1] = *(const v8h*)(Brow + k0 + 16 + 8 * half);
        c.v = __builtin_amdgcn_wmma_f32_16x16x32_f16(false, a.v, false, bb.v,
                                                     (short)0, c.v, false, false);
    }

    if (n < H_) {
        for (int i = 0; i < 8; ++i) {
            int u  = mtile * 16 + 8 * half + i;     // D: M = i + 8*(lane/16)
            int gi = (b * U_ + u) * H_ + n;
            float x = sigmoidf_(uq[gi]) * usv_embed[gi] + c.f[i];
            float o = eluf_(x);
            out_usv[gi] = o;
            usvuT[((size_t)b * H_ + n) * U_ + u] = (_Float16)o;  // transposed f16 for K8
        }
    }
}

// ---------------------------------------------------------------------------
// K4: uk = usv_updated @ W_Q + b_Q
// ---------------------------------------------------------------------------
__global__ void k_uk(const float* __restrict__ out_usv, const float* __restrict__ W_Q,
                     const float* __restrict__ b_Q, float* __restrict__ uk)
{
    int i = blockIdx.x * blockDim.x + threadIdx.x;   // b*U+u
    if (i >= B_ * U_) return;
    float x[H_];
    for (int h = 0; h < H_; ++h) x[h] = out_usv[(size_t)i * H_ + h];
    for (int h = 0; h < H_; ++h) {
        float a = b_Q[h];
        for (int g = 0; g < H_; ++g) a += x[g] * W_Q[g * H_ + h];
        uk[(size_t)i * H_ + h] = a;
    }
}

// ---------------------------------------------------------------------------
// K5: 9 nearest tasks per (b,t) by squared 2D distance (order-equivalent to
// sqrt); stable lowest-index tiebreak matches jax.lax.top_k. Wave per row.
// ---------------------------------------------------------------------------
__global__ void k_topk(const float* __restrict__ task_f, int* __restrict__ nbr)
{
    int lane = threadIdx.x & 31;
    int row  = blockIdx.x * (blockDim.x >> 5) + (threadIdx.x >> 5); // b*T+t
    if (row >= B_ * T_) return;
    int b = row / T_;
    float cx = task_f[(size_t)row * 5 + 0];
    float cy = task_f[(size_t)row * 5 + 1];
    const float* tf = task_f + (size_t)b * T_ * 5;

    float d[16];
    for (int it = 0; it < 16; ++it) {
        int t2 = lane + 32 * it;
        float dx = cx - tf[(size_t)t2 * 5 + 0];
        float dy = cy - tf[(size_t)t2 * 5 + 1];
        d[it] = dx * dx + dy * dy;
    }
    unsigned sel = 0;
    for (int j = 0; j < KNN; ++j) {
        float bv = 3.4e38f;
        int   bi = 1 << 30;
        for (int it = 0; it < 16; ++it) {
            if (!((sel >> it) & 1u)) {
                int t2 = lane + 32 * it;
                if (d[it] < bv || (d[it] == bv && t2 < bi)) { bv = d[it]; bi = t2; }
            }
        }
        for (int m = 16; m >= 1; m >>= 1) {
            float ov = __shfl_xor(bv, m, 32);
            int   oi = __shfl_xor(bi, m, 32);
            if (ov < bv || (ov == bv && oi < bi)) { bv = ov; bi = oi; }
        }
        if ((bi & 31) == lane) sel |= 1u << (bi >> 5);
        if (lane == 0) nbr[(size_t)row * KNN + j] = bi;
    }
}

// ---------------------------------------------------------------------------
// K6: task<-task attention over the 9-neighbor mask only (exact: masked
// entries softmax to ~0 at NEG). Thread per (b,t). agg_tasks -> ws.
// ---------------------------------------------------------------------------
__global__ void k_task_nbr(const float* __restrict__ tq, const float* __restrict__ tk,
                           const float* __restrict__ te, const int* __restrict__ nbr,
                           const float* __restrict__ W_at, const float* __restrict__ b_at,
                           float* __restrict__ aggt)
{
    int row = blockIdx.x * blockDim.x + threadIdx.x;  // b*T+t
    if (row >= B_ * T_) return;
    int b = row / T_;
    float q[H_], wat[H_];
    for (int h = 0; h < H_; ++h) { q[h] = tq[(size_t)row * H_ + h]; wat[h] = W_at[h]; }
    float bat = b_at[0];

    int id[KNN]; float s[KNN];
    float mx = -3.4e38f;
    for (int j = 0; j < KNN; ++j) {
        int sI = nbr[(size_t)row * KNN + j];
        id[j] = sI;
        const float* tks = tk + (size_t)(b * T_ + sI) * H_;
        float acc = bat;
        for (int h = 0; h < H_; ++h) acc += tanhf(q[h] + tks[h]) * wat[h];
        s[j] = acc;
        mx = fmaxf(mx, acc);
    }
    float sum = 0.0f;
    for (int j = 0; j < KNN; ++j) { s[j] = expf(s[j] - mx); sum += s[j]; }
    float inv = 1.0f / sum;
    float acc[H_] = {0.f, 0.f, 0.f, 0.f, 0.f, 0.f, 0.f, 0.f};
    for (int j = 0; j < KNN; ++j) {
        const float* tes = te + (size_t)(b * T_ + id[j]) * H_;
        float p = s[j] * inv;
        for (int h = 0; h < H_; ++h) acc[h] += p * tes[h];
    }
    for (int h = 0; h < H_; ++h) aggt[(size_t)row * H_ + h] = acc[h];
}

// ---------------------------------------------------------------------------
// K7: tu attention, softmax over U per (b,t). Wave per (b,t), 2 u's per lane.
// Stores tuw transposed [b][t][u] as f16 (WMMA A operand for K8).
// ---------------------------------------------------------------------------
__global__ void k_tu_attn(const float* __restrict__ uk, const float* __restrict__ tq,
                          const float* __restrict__ W_at, const float* __restrict__ b_at,
                          _Float16* __restrict__ tuwT)
{
    int lane = threadIdx.x & 31;
    int row  = blockIdx.x * (blockDim.x >> 5) + (threadIdx.x >> 5);  // b*T+t
    if (row >= B_ * T_) return;
    int b = row / T_;
    float q[H_], wat[H_];
    for (int h = 0; h < H_; ++h) { q[h] = tq[(size_t)row * H_ + h]; wat[h] = W_at[h]; }
    float bat = b_at[0];

    const float* ukb = uk + (size_t)b * U_ * H_;
    int u0 = lane, u1 = lane + 32;
    float s0 = bat, s1 = bat;
    for (int h = 0; h < H_; ++h) {
        s0 += tanhf(ukb[(size_t)u0 * H_ + h] + q[h]) * wat[h];
        s1 += tanhf(ukb[(size_t)u1 * H_ + h] + q[h]) * wat[h];
    }
    float mx = fmaxf(s0, s1);
    for (int m = 16; m >= 1; m >>= 1) mx = fmaxf(mx, __shfl_xor(mx, m, 32));
    float p0 = expf(s0 - mx), p1 = expf(s1 - mx);
    float sum = p0 + p1;
    for (int m = 16; m >= 1; m >>= 1) sum += __shfl_xor(sum, m, 32);
    float inv = 1.0f / sum;
    tuwT[(size_t)row * U_ + u0] = (_Float16)(p0 * inv);
    tuwT[(size_t)row * U_ + u1] = (_Float16)(p1 * inv);
}

// ---------------------------------------------------------------------------
// K8: agg_usv = tuw^T(512x64) @ usv_updated(64x8) via WMMA; fused epilogue:
// task_updated = elu(sig(tq)*task_embed + agg_tasks + agg_usv).
// Same n&7 column-duplication trick as K3: no divergence in the K loop.
// ---------------------------------------------------------------------------
__global__ void k_aggusv_wmma(const _Float16* __restrict__ tuwT, const _Float16* __restrict__ usvuT,
                              const float* __restrict__ tq, const float* __restrict__ te,
                              const float* __restrict__ aggt, float* __restrict__ out_task)
{
    int lane = threadIdx.x & 31;
    int row  = blockIdx.x * (blockDim.x >> 5) + (threadIdx.x >> 5);  // b*32 + mtile
    int b     = row >> 5;
    int mtile = row & 31;
    int half  = lane >> 4;
    int n     = lane & 15;

    V16 a, bb;
    V8F c;
    for (int i = 0; i < 8; ++i) c.f[i] = 0.0f;

    const _Float16* Arow = tuwT + ((size_t)(b * T_) + mtile * 16 + n) * U_;
    const _Float16* Brow = usvuT + ((size_t)b * H_ + (n & 7)) * U_;

    for (int kk = 0; kk < U_ / 32; ++kk) {
        int k0 = kk * 32;
        a.h[0] = *(const v8h*)(Arow + k0 + 8 * half);
        a.h[1] = *(const v8h*)(Arow + k0 + 16 + 8 * half);
        bb.h[0] = *(const v8h*)(Brow + k0 + 8 * half);
        bb.h[1] = *(const v8h*)(Brow + k0 + 16 + 8 * half);
        c.v = __builtin_amdgcn_wmma_f32_16x16x32_f16(false, a.v, false, bb.v,
                                                     (short)0, c.v, false, false);
    }

    if (n < H_) {
        for (int i = 0; i < 8; ++i) {
            int t  = mtile * 16 + 8 * half + i;
            int gi = (b * T_ + t) * H_ + n;
            float x = sigmoidf_(tq[gi]) * te[gi] + aggt[gi] + c.f[i];
            out_task[gi] = eluf_(x);
        }
    }
}

// ---------------------------------------------------------------------------
extern "C" void kernel_launch(void* const* d_in, const int* in_sizes, int n_in,
                              void* d_out, int out_size, void* d_ws, size_t ws_size,
                              hipStream_t stream)
{
    const float* usv_f  = (const float*)d_in[0];
    const float* task_f = (const float*)d_in[1];
    const int*   adj    = (const int*)d_in[2];
    // d_in[3] = edge_features: unused by the reference math.
    const float* W_usv  = (const float*)d_in[4];
    const float* b_usv  = (const float*)d_in[5];
    const float* W_task = (const float*)d_in[6];
    const float* b_task = (const float*)d_in[7];
    const float* W_au   = (const float*)d_in[8];
    const float* b_au   = (const float*)d_in[9];
    const float* W_at   = (const float*)d_in[10];
    const float* b_at   = (const float*)d_in[11];
    const float* W_Q    = (const float*)d_in[12];
    const float* b_Q    = (const float*)d_in[13];
    const float* W_K    = (const float*)d_in[14];
    const float* b_K    = (const float*)d_in[15];
    const float* W_V    = (const float*)d_in[16];
    const float* b_V    = (const float*)d_in[17];

    float* out_usv  = (float*)d_out;                        // (B,U,H)
    float* out_task = (float*)d_out + (size_t)B_ * U_ * H_; // (B,T,H)

    // workspace carve-up (all regions 16B aligned); ~14.1 MB total
    char* p = (char*)d_ws;
    float* task_embed = (float*)p;  p += (size_t)B_ * T_ * H_ * 4;
    float* tk         = (float*)p;  p += (size_t)B_ * T_ * H_ * 4;
    float* tq         = (float*)p;  p += (size_t)B_ * T_ * H_ * 4;
    float* aggt       = (float*)p;  p += (size_t)B_ * T_ * H_ * 4;
    float* usv_embed  = (float*)p;  p += (size_t)B_ * U_ * H_ * 4;
    float* uq         = (float*)p;  p += (size_t)B_ * U_ * H_ * 4;
    float* uk         = (float*)p;  p += (size_t)B_ * U_ * H_ * 4;
    _Float16* teT     = (_Float16*)p; p += (size_t)B_ * H_ * T_ * 2;
    _Float16* w16     = (_Float16*)p; p += (size_t)B_ * U_ * T_ * 2;
    _Float16* tuwT    = (_Float16*)p; p += (size_t)B_ * T_ * U_ * 2;
    _Float16* usvuT   = (_Float16*)p; p += (size_t)B_ * H_ * U_ * 2;
    int* nbr          = (int*)p;      p += (size_t)B_ * T_ * KNN * 4;

    int tot1 = B_ * T_ + B_ * U_;
    k_embed<<<(tot1 + 255) / 256, 256, 0, stream>>>(
        usv_f, task_f, W_usv, b_usv, W_task, b_task, W_Q, b_Q, W_K, b_K, W_V, b_V,
        task_embed, tk, tq, usv_embed, uq, teT);

    k_usv_attn<<<(B_ * U_) / 8, 256, 0, stream>>>(uq, tk, adj, W_au, b_au, w16);

    k_aggtask_wmma<<<B_, 128, 0, stream>>>(w16, teT, uq, usv_embed, out_usv, usvuT);

    k_uk<<<(B_ * U_ + 255) / 256, 256, 0, stream>>>(out_usv, W_Q, b_Q, uk);

    k_topk<<<(B_ * T_) / 8, 256, 0, stream>>>(task_f, nbr);

    k_task_nbr<<<(B_ * T_ + 255) / 256, 256, 0, stream>>>(
        tq, tk, task_embed, nbr, W_at, b_at, aggt);

    k_tu_attn<<<(B_ * T_) / 8, 256, 0, stream>>>(uk, tq, W_at, b_at, tuwT);

    k_aggusv_wmma<<<(B_ * (T_ / 16)) / 8, 256, 0, stream>>>(
        tuwT, usvuT, tq, task_embed, aggt, out_task);
}
